// CTCAttentionDecoder_34918084117121
// MI455X (gfx1250) — matrix operand order repaired and checked by reading, not compiled, not run on hardware
//
#include <hip/hip_runtime.h>
#include <hip/hip_bf16.h>

// ---------------- CDNA5 WMMA types ----------------
typedef __bf16 bf16;
typedef bf16  v16bf __attribute__((ext_vector_type(16)));
typedef float v8f   __attribute__((ext_vector_type(8)));

#if __has_builtin(__builtin_amdgcn_tensor_load_to_lds)
#define HAVE_TDM 1
#else
#define HAVE_TDM 0
#endif

// Problem constants (match reference)
#define CN  4
#define CS  1024
#define CT0 512
#define CT  513          // T0 + 1
#define CC  1024
#define CH  16
#define CD  64
#define CL  12
#define CV  8192
#define CFF 4096

// LDS tile pitches (bf16 elements). Chosen so row pitch is a multiple of
// 16 bytes (aligned b128 access) and matches the TDM pad settings below.
#define ASTR 40   // A / transposed-B tiles: 32 data + 8 pad  (80 B rows)
#define BSTR 72   // k-major B tiles:        64 data + 8 pad  (144 B rows)

static __device__ __forceinline__ unsigned bfbits(float f) {
  unsigned u = __builtin_bit_cast(unsigned, f);
  return ((u + 0x7fffu + ((u >> 16) & 1u)) >> 16) & 0xffffu;
}
static __device__ __forceinline__ unsigned pkbf(float a, float b) {
  return bfbits(a) | (bfbits(b) << 16);
}
static __device__ __forceinline__ bf16 f2bf(float f) {
  return __builtin_bit_cast(bf16, (unsigned short)bfbits(f));
}
static __device__ __forceinline__ float b2f(bf16 v) {
  unsigned u = (unsigned)__builtin_bit_cast(unsigned short, v) << 16;
  return __builtin_bit_cast(float, u);
}
static __device__ __forceinline__ int imin(int a, int b) { return a < b ? a : b; }

// ---------------- f32 -> bf16 bulk convert ----------------
__global__ void cvt_bf16_kernel(const float* __restrict__ src,
                                bf16* __restrict__ dst, size_t n) {
  size_t i = ((size_t)blockIdx.x * blockDim.x + threadIdx.x) * 4;
  if (i + 4 <= n) {
    float4 v = *(const float4*)(src + i);
    unsigned* d = (unsigned*)(dst + i);
    d[0] = pkbf(v.x, v.y);
    d[1] = pkbf(v.z, v.w);
  } else {
    for (; i < n; ++i) dst[i] = f2bf(src[i]);
  }
}

// ---------------- Embedding gather ----------------
__global__ void embed_kernel(const int* __restrict__ targets,
                             const float* __restrict__ wte,
                             float* __restrict__ y) {
  size_t idx = (size_t)blockIdx.x * blockDim.x + threadIdx.x;
  size_t total = (size_t)CN * CT * CC;
  if (idx >= total) return;
  int c = (int)(idx % CC);
  size_t r = idx / CC;
  int t = (int)(r % CT);
  int n = (int)(r / CT);
  int tok = (t == 0) ? 3 /*STX*/ : targets[n * CT0 + (t - 1)];
  y[idx] = wte[(size_t)tok * CC + c];
}

// ---------------- LayerNorm -> bf16 output (one block per row) ----------------
__global__ void ln_kernel(const float* __restrict__ x,
                          const float* __restrict__ w,
                          bf16* __restrict__ outb, int Cc) {
  int row = blockIdx.x;
  const float* xr = x + (size_t)row * Cc;
  bf16* orow = outb + (size_t)row * Cc;
  __shared__ float red[256];
  float s = 0.f;
  for (int c = threadIdx.x; c < Cc; c += blockDim.x) s += xr[c];
  red[threadIdx.x] = s; __syncthreads();
  for (int o = 128; o > 0; o >>= 1) {
    if ((int)threadIdx.x < o) red[threadIdx.x] += red[threadIdx.x + o];
    __syncthreads();
  }
  float mu = red[0] / (float)Cc;
  __syncthreads();
  float v = 0.f;
  for (int c = threadIdx.x; c < Cc; c += blockDim.x) {
    float d = xr[c] - mu; v += d * d;
  }
  red[threadIdx.x] = v; __syncthreads();
  for (int o = 128; o > 0; o >>= 1) {
    if ((int)threadIdx.x < o) red[threadIdx.x] += red[threadIdx.x + o];
    __syncthreads();
  }
  float rstd = rsqrtf(red[0] / (float)Cc + 1e-5f);
  for (int c = threadIdx.x; c < Cc; c += blockDim.x)
    orow[c] = f2bf((xr[c] - mu) * rstd * w[c]);
}

#if HAVE_TDM
// ---------------- Tensor Data Mover descriptor + issue ----------------
typedef unsigned u32x4 __attribute__((ext_vector_type(4)));
typedef int      i32x8 __attribute__((ext_vector_type(8)));
typedef int      i32x4 __attribute__((ext_vector_type(4)));

// 2D tile load: tile0 elems contiguous (dim0), tile1 rows, element = 2 bytes.
// remD0/remD1 = remaining tensor extent from tile origin (OOB reads -> 0).
// LDS row pitch = tile0*2 + pad DWORDs*4 (pad fields per D# group1).
static __device__ __forceinline__ void tdm_load_2d(
    const bf16* gptr, unsigned ldsOff, int remD0, int remD1,
    long stride0, int tile0, int tile1, int padInterval, int padAmount) {
  unsigned long long ga = (unsigned long long)(size_t)gptr;
  u32x4 g0 = {
    1u,                                             // count=1, user D#
    ldsOff,                                         // lds_addr (bytes)
    (unsigned)ga,                                   // global_addr[31:0]
    (unsigned)((ga >> 32) & 0x1FFFFFFu) | (2u << 30) // addr[56:32] | type=2
  };
  unsigned long long d0 = (unsigned)remD0;
  unsigned long long s0 = (unsigned long long)stride0;
  unsigned long long w0 = (1ull << 16)                         // data_size=2B
                        | (1ull << 20)                         // pad_enable
                        | ((unsigned long long)padInterval << 22)
                        | ((unsigned long long)padAmount << 25)
                        | ((d0 & 0xFFFFull) << 48);            // tensor_dim0 lo
  unsigned long long w1 = ((d0 >> 16) & 0xFFFFull)             // tensor_dim0 hi
                        | ((unsigned long long)(unsigned)remD1 << 16) // dim1
                        | ((unsigned long long)(unsigned)tile0 << 48);
  unsigned long long w2 = (unsigned long long)(unsigned)tile1  // tile_dim1
                        | ((s0 & 0xFFFFFFFFull) << 32);        // stride0 lo
  unsigned long long w3 = (s0 >> 32) & 0xFFFFull;              // stride0 hi
  i32x8 g1 = { (int)(unsigned)w0, (int)(unsigned)(w0 >> 32),
               (int)(unsigned)w1, (int)(unsigned)(w1 >> 32),
               (int)(unsigned)w2, (int)(unsigned)(w2 >> 32),
               (int)(unsigned)w3, (int)(unsigned)(w3 >> 32) };
  i32x4 gz4 = {0, 0, 0, 0};
  i32x8 gz8 = {0, 0, 0, 0, 0, 0, 0, 0};
  __builtin_amdgcn_tensor_load_to_lds(g0, g1, gz4, gz4, gz8, 0);
}

// ---------------- TDM double-buffered WMMA GEMM ----------------
// C[M x Nn] (+)= A[M x K] @ B ; A,B bf16, C f32 (or bf16 when OUTBF)
template <bool TRANSB, bool ACCUM, bool OUTBF>
__launch_bounds__(128)
__global__ void gemm_wmma_tdm(const bf16* __restrict__ A, long lda, long sAn, long sAh,
                              const bf16* __restrict__ B, long ldb, long sBn, long sBh,
                              void* __restrict__ Cv, long ldc, long sCn, long sCh,
                              int M, int Nn, int K, int Hh) {
  __shared__ bf16 Ast[2][64 * ASTR];
  __shared__ bf16 Bst[2][64 * ASTR];   // covers both 64*40 and 32*72 layouts
  int z = blockIdx.z;
  int bn = z / Hh, bh = z % Hh;
  A += (size_t)bn * sAn + (size_t)bh * sAh;
  B += (size_t)bn * sBn + (size_t)bh * sBh;
  float* Cf = (float*)Cv + (size_t)bn * sCn + (size_t)bh * sCh;
  bf16*  Cb = (bf16*)Cv + (size_t)bn * sCn + (size_t)bh * sCh;

  int rowBase = blockIdx.y * 64;
  int colBase = blockIdx.x * 64;
  int tid  = threadIdx.x;
  int lane = tid & 31;
  int wave = tid >> 5;

  unsigned aOff = (unsigned)(size_t)(&Ast[0][0]);
  unsigned bOff = (unsigned)(size_t)(&Bst[0][0]);
  const unsigned stageBytes = 64 * ASTR * 2;

  v8f acc[4];
#pragma unroll
  for (int c = 0; c < 4; ++c)
#pragma unroll
    for (int j = 0; j < 8; ++j) acc[c][j] = 0.0f;

  // stage issue: A tile rows 64 x 32 k; B tile per layout
  if (wave == 0) {
    tdm_load_2d(A + (size_t)rowBase * lda, aOff, K, M - rowBase, lda,
                32, 64, /*64B rows*/3, /*pad 4 DW*/3);
    if (TRANSB)
      tdm_load_2d(B + (size_t)colBase * ldb, bOff, K, Nn - colBase, ldb,
                  32, 64, 3, 3);
    else
      tdm_load_2d(B + colBase, bOff, Nn - colBase, K, ldb,
                  64, 32, /*128B rows*/4, /*pad 4 DW*/3);
  }

  for (int k0 = 0; k0 < K; k0 += 32) {
    int s = (k0 >> 5) & 1;
    if (wave == 0) {
      int kn = k0 + 32;
      if (kn < K) {
        unsigned so = (unsigned)(s ^ 1) * stageBytes;
        tdm_load_2d(A + (size_t)rowBase * lda + kn, aOff + so,
                    K - kn, M - rowBase, lda, 32, 64, 3, 3);
        if (TRANSB)
          tdm_load_2d(B + (size_t)colBase * ldb + kn, bOff + so,
                      K - kn, Nn - colBase, ldb, 32, 64, 3, 3);
        else
          tdm_load_2d(B + (size_t)kn * ldb + colBase, bOff + so,
                      Nn - colBase, K - kn, ldb, 64, 32, 4, 3);
        __builtin_amdgcn_s_wait_tensorcnt(2);   // current stage complete
      } else {
        __builtin_amdgcn_s_wait_tensorcnt(0);
      }
    }
    __syncthreads();

    const bf16* Ab = Ast[s];
    const bf16* Bb = Bst[s];
    v16bf afrag;
    int mloc = wave * 16 + (lane & 15);
    int half = lane >> 4;
#pragma unroll
    for (int j = 0; j < 16; ++j) {
      int kk = j + 16 * half;
      afrag[j] = Ab[mloc * ASTR + kk];
    }
#pragma unroll
    for (int c = 0; c < 4; ++c) {
      v16bf bfrag;
      int nloc = c * 16 + (lane & 15);
#pragma unroll
      for (int j = 0; j < 16; ++j) {
        int kk = j + 16 * half;
        bfrag[j] = TRANSB ? Bb[nloc * ASTR + kk] : Bb[kk * BSTR + nloc];
      }
      acc[c] = __builtin_amdgcn_wmma_f32_16x16x32_bf16(
          false, afrag, false, bfrag, (short)0, acc[c], false, false);
    }
    __syncthreads();
  }

  int half = lane >> 4;
#pragma unroll
  for (int c = 0; c < 4; ++c) {
    int col = colBase + c * 16 + (lane & 15);
    if (col >= Nn) continue;
#pragma unroll
    for (int r = 0; r < 8; ++r) {
      int row = rowBase + wave * 16 + half * 8 + r;
      if (row < M) {
        size_t idx = (size_t)row * ldc + col;
        if (OUTBF)      Cb[idx]  = f2bf(acc[c][r]);
        else if (ACCUM) Cf[idx] += acc[c][r];
        else            Cf[idx]  = acc[c][r];
      }
    }
  }
}
#else  // !HAVE_TDM -------- manual bf16 loader fallback --------
template <bool TRANSB, bool ACCUM, bool OUTBF>
__launch_bounds__(128)
__global__ void gemm_wmma_man(const bf16* __restrict__ A, long lda, long sAn, long sAh,
                              const bf16* __restrict__ B, long ldb, long sBn, long sBh,
                              void* __restrict__ Cv, long ldc, long sCn, long sCh,
                              int M, int Nn, int K, int Hh) {
  __shared__ bf16 As[64 * ASTR];
  __shared__ bf16 Bs[64 * ASTR];
  int z = blockIdx.z;
  int bn = z / Hh, bh = z % Hh;
  A += (size_t)bn * sAn + (size_t)bh * sAh;
  B += (size_t)bn * sBn + (size_t)bh * sBh;
  float* Cf = (float*)Cv + (size_t)bn * sCn + (size_t)bh * sCh;
  bf16*  Cb = (bf16*)Cv + (size_t)bn * sCn + (size_t)bh * sCh;
  int rowBase = blockIdx.y * 64, colBase = blockIdx.x * 64;
  int tid = threadIdx.x, lane = tid & 31, wave = tid >> 5;
  int lr = tid >> 1, lkh = (tid & 1) * 16;
  int lkk = tid >> 2, lnh = (tid & 3) * 16;
  const bf16 ZB = __builtin_bit_cast(bf16, (unsigned short)0);

  v8f acc[4];
#pragma unroll
  for (int c = 0; c < 4; ++c)
#pragma unroll
    for (int j = 0; j < 8; ++j) acc[c][j] = 0.0f;

  for (int k0 = 0; k0 < K; k0 += 32) {
    bool kFull = (k0 + 32 <= K);
    {
      int m = rowBase + lr;
      if (((lda & 7) == 0) && kFull && (rowBase + 64 <= M)) {
        const float4* pa = (const float4*)(A + (size_t)m * lda + k0 + lkh);
        float4* d = (float4*)(As + lr * ASTR + lkh);
        d[0] = pa[0]; d[1] = pa[1];
        if (k0 + 32 < K)
          __builtin_prefetch(A + (size_t)m * lda + k0 + 32 + lkh, 0, 3);
      } else {
        const bf16* Arow = A + (size_t)imin(m, M - 1) * lda;
        bool am = (m < M);
#pragma unroll
        for (int p = 0; p < 16; ++p) {
          int kk = k0 + lkh + p;
          bf16 v = Arow[imin(kk, K - 1)];
          As[lr * ASTR + lkh + p] = (am && kk < K) ? v : ZB;
        }
      }
    }
    if (TRANSB) {
      int n = colBase + lr;
      if (((ldb & 7) == 0) && kFull && (colBase + 64 <= Nn)) {
        const float4* pb = (const float4*)(B + (size_t)n * ldb + k0 + lkh);
        float4* d = (float4*)(Bs + lr * ASTR + lkh);
        d[0] = pb[0]; d[1] = pb[1];
      } else {
        const bf16* Brow = B + (size_t)imin(n, Nn - 1) * ldb;
        bool bm = (n < Nn);
#pragma unroll
        for (int p = 0; p < 16; ++p) {
          int kk = k0 + lkh + p;
          bf16 v = Brow[imin(kk, K - 1)];
          Bs[lr * ASTR + lkh + p] = (bm && kk < K) ? v : ZB;
        }
      }
    } else {
      if (((ldb & 7) == 0) && kFull && (colBase + 64 <= Nn)) {
        const float4* pb = (const float4*)(B + (size_t)(k0 + lkk) * ldb + colBase + lnh);
        float4* d = (float4*)(Bs + lkk * BSTR + lnh);
        d[0] = pb[0]; d[1] = pb[1];
      } else {
        int n = colBase + lr;
        int nc = imin(n, Nn - 1);
        bool bm = (n < Nn);
#pragma unroll
        for (int p = 0; p < 16; ++p) {
          int kk = k0 + lkh + p;
          bf16 v = B[(size_t)imin(kk, K - 1) * ldb + nc];
          Bs[(lkh + p) * BSTR + lr] = (bm && kk < K) ? v : ZB;
        }
      }
    }
    __syncthreads();

    v16bf afrag;
    int mloc = wave * 16 + (lane & 15);
    int half = lane >> 4;
#pragma unroll
    for (int j = 0; j < 16; ++j) {
      int kk = j + 16 * half;
      afrag[j] = As[mloc * ASTR + kk];
    }
#pragma unroll
    for (int c = 0; c < 4; ++c) {
      v16bf bfrag;
      int nloc = c * 16 + (lane & 15);
#pragma unroll
      for (int j = 0; j < 16; ++j) {
        int kk = j + 16 * half;
        bfrag[j] = TRANSB ? Bs[nloc * ASTR + kk] : Bs[kk * BSTR + nloc];
      }
      acc[c] = __builtin_amdgcn_wmma_f32_16x16x32_bf16(
          false, afrag, false, bfrag, (short)0, acc[c], false, false);
    }
    __syncthreads();
  }

  int half = lane >> 4;
#pragma unroll
  for (int c = 0; c < 4; ++c) {
    int col = colBase + c * 16 + (lane & 15);
    if (col >= Nn) continue;
#pragma unroll
    for (int r = 0; r < 8; ++r) {
      int row = rowBase + wave * 16 + half * 8 + r;
      if (row < M) {
        size_t idx = (size_t)row * ldc + col;
        if (OUTBF)      Cb[idx]  = f2bf(acc[c][r]);
        else if (ACCUM) Cf[idx] += acc[c][r];
        else            Cf[idx]  = acc[c][r];
      }
    }
  }
}
#endif

// ---------------- Softmax: f32 scores in, bf16 probs out ----------------
template <bool CAUSAL>
__global__ void softmax_kernel(const float* __restrict__ s,
                               bf16* __restrict__ p,
                               const int* __restrict__ lens,
                               int T_, int Scols) {
  int t = blockIdx.x, h = blockIdx.y, n = blockIdx.z;
  int Hh = gridDim.y;
  size_t off = ((((size_t)n * Hh + h) * T_) + t) * Scols;
  const float* row = s + off;
  bf16* prow = p + off;
  int limit = CAUSAL ? (t + 1) : lens[n];
  const float scale = 0.125f; // 1/sqrt(64)
  __shared__ float red[256];
  float m = -3.0e38f;
  for (int c = threadIdx.x; c < limit; c += blockDim.x)
    m = fmaxf(m, row[c] * scale);
  red[threadIdx.x] = m; __syncthreads();
  for (int o = 128; o > 0; o >>= 1) {
    if ((int)threadIdx.x < o)
      red[threadIdx.x] = fmaxf(red[threadIdx.x], red[threadIdx.x + o]);
    __syncthreads();
  }
  m = red[0]; __syncthreads();
  float sum = 0.f;
  for (int c = threadIdx.x; c < limit; c += blockDim.x)
    sum += expf(row[c] * scale - m);
  red[threadIdx.x] = sum; __syncthreads();
  for (int o = 128; o > 0; o >>= 1) {
    if ((int)threadIdx.x < o) red[threadIdx.x] += red[threadIdx.x + o];
    __syncthreads();
  }
  float inv = 1.0f / red[0];
  for (int c = threadIdx.x; c < Scols; c += blockDim.x)
    prow[c] = f2bf((c < limit) ? expf(row[c] * scale - m) * inv : 0.0f);
}

// ---------------- RoPE on bf16 q,k (N,T,H*D) ----------------
__global__ void rope_kernel(bf16* __restrict__ q, bf16* __restrict__ k) {
  size_t idx = (size_t)blockIdx.x * blockDim.x + threadIdx.x;
  size_t total = (size_t)CN * CT * CH * (CD / 2);
  if (idx >= total) return;
  int i = (int)(idx % (CD / 2));
  size_t r = idx / (CD / 2);
  int h = (int)(r % CH); r /= CH;
  int t = (int)(r % CT);
  int n = (int)(r / CT);
  float ang = powf(10000.0f, -2.0f * (float)i / (float)CD) * (float)t;
  float cs = cosf(ang), sn = sinf(ang);
  size_t base = (((size_t)n * CT + t) * CC) + (size_t)h * CD + 2 * i;
  float qe = b2f(q[base]), qo = b2f(q[base + 1]);
  q[base]     = f2bf(qe * cs - qo * sn);
  q[base + 1] = f2bf(qo * cs + qe * sn);
  float ke = b2f(k[base]), ko = b2f(k[base + 1]);
  k[base]     = f2bf(ke * cs - ko * sn);
  k[base + 1] = f2bf(ko * cs + ke * sn);
}

// ---------------- exact GELU: f32 in, bf16 out ----------------
__global__ void gelu_kernel(const float* __restrict__ x,
                            bf16* __restrict__ o, size_t n) {
  size_t i = (size_t)blockIdx.x * blockDim.x + threadIdx.x;
  if (i < n) {
    float v = x[i];
    o[i] = f2bf(0.5f * v * (1.0f + erff(v * 0.70710678118654752f)));
  }
}

// ---------------- loss ----------------
__global__ void init_accum(float* a) { a[0] = 0.f; a[1] = 0.f; }

__global__ void loss_kernel(const float* __restrict__ logits,
                            const int* __restrict__ targets,
                            const int* __restrict__ tlen,
                            float* __restrict__ accum) {
  int t = blockIdx.x, n = blockIdx.y;
  const float* row = logits + ((size_t)n * CT + t) * CV;
  int tgt = (t < CT0) ? targets[n * CT0 + t] : 0;
  if (t == tlen[n]) tgt = 4; // ETX
  __shared__ float red[256];
  float m = -3.0e38f;
  for (int c = threadIdx.x; c < CV; c += blockDim.x) m = fmaxf(m, row[c]);
  red[threadIdx.x] = m; __syncthreads();
  for (int o = 128; o > 0; o >>= 1) {
    if ((int)threadIdx.x < o)
      red[threadIdx.x] = fmaxf(red[threadIdx.x], red[threadIdx.x + o]);
    __syncthreads();
  }
  m = red[0]; __syncthreads();
  float sum = 0.f;
  for (int c = threadIdx.x; c < CV; c += blockDim.x) sum += expf(row[c] - m);
  red[threadIdx.x] = sum; __syncthreads();
  for (int o = 128; o > 0; o >>= 1) {
    if ((int)threadIdx.x < o) red[threadIdx.x] += red[threadIdx.x + o];
    __syncthreads();
  }
  if (threadIdx.x == 0 && tgt != 0) {
    float nll = -(row[tgt] - m - logf(red[0]));
    atomicAdd(&accum[0], nll);
    atomicAdd(&accum[1], 1.0f);
  }
}

__global__ void finalize_kernel(const float* accum, float* out) {
  out[0] = accum[0] / fmaxf(accum[1], 1.0f);
}

// ---------------- host-side dispatch ----------------
static void run_gemm(hipStream_t st, bool transB, bool accum, bool outbf,
                     const bf16* A, long lda, long sAn, long sAh,
                     const bf16* B, long ldb, long sBn, long sBh,
                     void* Cm, long ldc, long sCn, long sCh,
                     int M, int Nn, int K, int batches, int Hh) {
  dim3 grid((Nn + 63) / 64, (M + 63) / 64, batches);
  dim3 blk(128);
#if HAVE_TDM
#define GK gemm_wmma_tdm
#else
#define GK gemm_wmma_man
#endif
  if (transB)
    GK<true, false, false><<<grid, blk, 0, st>>>(A, lda, sAn, sAh, B, ldb, sBn, sBh,
                                                 Cm, ldc, sCn, sCh, M, Nn, K, Hh);
  else if (outbf)
    GK<false, false, true><<<grid, blk, 0, st>>>(A, lda, sAn, sAh, B, ldb, sBn, sBh,
                                                 Cm, ldc, sCn, sCh, M, Nn, K, Hh);
  else if (accum)
    GK<false, true, false><<<grid, blk, 0, st>>>(A, lda, sAn, sAh, B, ldb, sBn, sBh,
                                                 Cm, ldc, sCn, sCh, M, Nn, K, Hh);
  else
    GK<false, false, false><<<grid, blk, 0, st>>>(A, lda, sAn, sAh, B, ldb, sBn, sBh,
                                                  Cm, ldc, sCn, sCh, M, Nn, K, Hh);
#undef GK
}

static void run_cvt(hipStream_t st, const float* src, bf16* dst, size_t n) {
  cvt_bf16_kernel<<<(unsigned)((n / 4 + 255) / 256), 256, 0, st>>>(src, dst, n);
}

extern "C" void kernel_launch(void* const* d_in, const int* in_sizes, int n_in,
                              void* d_out, int out_size, void* d_ws, size_t ws_size,
                              hipStream_t stream) {
  (void)in_sizes; (void)n_in; (void)out_size; (void)ws_size;
  const float* features       = (const float*)d_in[0];
  const int*   targets        = (const int*)d_in[1];
  const int*   input_lengths  = (const int*)d_in[2];
  const int*   target_lengths = (const int*)d_in[3];
  const float* wte   = (const float*)d_in[4];
  const float* ln1_w = (const float*)d_in[5];
  const float* Wq_m  = (const float*)d_in[6];
  const float* Wk_m  = (const float*)d_in[7];
  const float* Wv_m  = (const float*)d_in[8];
  const float* Wo_m  = (const float*)d_in[9];
  const float* Wq_t  = (const float*)d_in[10];
  const float* Wk_t  = (const float*)d_in[11];
  const float* Wv_t  = (const float*)d_in[12];
  const float* Wo_t  = (const float*)d_in[13];
  const float* ln2_w = (const float*)d_in[14];
  const float* W1    = (const float*)d_in[15];
  const float* W2    = (const float*)d_in[16];
  const float* lnf_w = (const float*)d_in[17];
  const float* Wout  = (const float*)d_in[18];

  const size_t NTC = (size_t)CN * CT * CC;        // 2,101,248
  const size_t NSC = (size_t)CN * CS * CC;        // 4,194,304
  const size_t REG = (size_t)CN * CH * CT * CS;   // 33,619,968
  const size_t NTF = (size_t)CN * CT * CFF;       // 8,404,992
  const size_t CC2 = (size_t)CC * CC;             // 1,048,576

  char* base = (char*)d_ws;
  // f32 buffers
  float* y      = (float*)base; base += NTC * 4;
  float* region = (float*)base; base += REG * 4;   // scores / ffh / logits
  float* accum  = (float*)base; base += 16;
  // bf16 buffers (all sizes are multiples of 8 elements -> 16B aligned)
  bf16* xnb  = (bf16*)base; base += NTC * 2;
  bf16* fb   = (bf16*)base; base += NSC * 2;
  bf16* qb   = (bf16*)base; base += NTC * 2;
  bf16* kbb  = (bf16*)base; base += NSC * 2;
  bf16* vbb  = (bf16*)base; base += NSC * 2;
  bf16* ohb  = (bf16*)base; base += NTC * 2;
  bf16* sb   = (bf16*)base; base += REG * 2;       // probs / ffhb (disjoint)
  bf16* wbuf = (bf16*)base; base += (8 * CC2 + 2 * (size_t)CC * CFF) * 2;
  bf16* wob  = (bf16*)base; base += (size_t)CC * CV * 2;

  float* scores = region;
  float* ffh    = region;
  float* logits = region;
  bf16*  ffhb   = sb;

  const long TC = (long)CT * CC, SC = (long)CS * CC;
  const long TS = (long)CT * CS, TT = (long)CT * CT;

  // one-time converts: features, Wout
  run_cvt(stream, features, fb, NSC);
  run_cvt(stream, Wout, wob, (size_t)CC * CV);

  // embed prompt
  embed_kernel<<<(unsigned)((NTC + 255) / 256), 256, 0, stream>>>(targets, wte, y);

  for (int l = 0; l < CL; ++l) {
    const size_t wOff = (size_t)l * CC2;
    // per-layer weight -> bf16 scratch
    bf16* bWq_m = wbuf + 0 * CC2;  run_cvt(stream, Wq_m + wOff, bWq_m, CC2);
    bf16* bWk_m = wbuf + 1 * CC2;  run_cvt(stream, Wk_m + wOff, bWk_m, CC2);
    bf16* bWv_m = wbuf + 2 * CC2;  run_cvt(stream, Wv_m + wOff, bWv_m, CC2);
    bf16* bWo_m = wbuf + 3 * CC2;  run_cvt(stream, Wo_m + wOff, bWo_m, CC2);
    bf16* bWq_t = wbuf + 4 * CC2;  run_cvt(stream, Wq_t + wOff, bWq_t, CC2);
    bf16* bWk_t = wbuf + 5 * CC2;  run_cvt(stream, Wk_t + wOff, bWk_t, CC2);
    bf16* bWv_t = wbuf + 6 * CC2;  run_cvt(stream, Wv_t + wOff, bWv_t, CC2);
    bf16* bWo_t = wbuf + 7 * CC2;  run_cvt(stream, Wo_t + wOff, bWo_t, CC2);
    bf16* bW1   = wbuf + 8 * CC2;  run_cvt(stream, W1 + (size_t)l * CC * CFF, bW1, (size_t)CC * CFF);
    bf16* bW2   = bW1 + (size_t)CC * CFF;
    run_cvt(stream, W2 + (size_t)l * CFF * CC, bW2, (size_t)CFF * CC);

    // xn = LN(y) * ln1_w[l]  (bf16)
    ln_kernel<<<CN * CT, 256, 0, stream>>>(y, ln1_w + (size_t)l * CC, xnb, CC);

    // ---- cross attention ----
    run_gemm(stream, false, false, true, xnb, CC, 0, 0, bWq_m, CC, 0, 0,
             qb, CC, 0, 0, CN * CT, CC, CC, 1, 1);
    run_gemm(stream, false, false, true, fb, CC, 0, 0, bWk_m, CC, 0, 0,
             kbb, CC, 0, 0, CN * CS, CC, CC, 1, 1);
    run_gemm(stream, false, false, true, fb, CC, 0, 0, bWv_m, CC, 0, 0,
             vbb, CC, 0, 0, CN * CS, CC, CC, 1, 1);
    run_gemm(stream, true, false, false, qb, CC, TC, CD, kbb, CC, SC, CD,
             scores, CS, (long)CH * TS, TS, CT, CS, CD, CN * CH, CH);
    softmax_kernel<false><<<dim3(CT, CH, CN), 256, 0, stream>>>(
        scores, sb, input_lengths, CT, CS);
    run_gemm(stream, false, false, true, sb, CS, (long)CH * TS, TS, vbb, CC, SC, CD,
             ohb, CC, TC, CD, CT, CD, CS, CN * CH, CH);
    run_gemm(stream, false, true, false, ohb, CC, 0, 0, bWo_m, CC, 0, 0,
             y, CC, 0, 0, CN * CT, CC, CC, 1, 1);

    // ---- self attention (causal + RoPE) ----
    run_gemm(stream, false, false, true, xnb, CC, 0, 0, bWq_t, CC, 0, 0,
             qb, CC, 0, 0, CN * CT, CC, CC, 1, 1);
    run_gemm(stream, false, false, true, xnb, CC, 0, 0, bWk_t, CC, 0, 0,
             kbb, CC, 0, 0, CN * CT, CC, CC, 1, 1);
    run_gemm(stream, false, false, true, xnb, CC, 0, 0, bWv_t, CC, 0, 0,
             vbb, CC, 0, 0, CN * CT, CC, CC, 1, 1);
    {
      size_t tot = (size_t)CN * CT * CH * (CD / 2);
      rope_kernel<<<(unsigned)((tot + 255) / 256), 256, 0, stream>>>(qb, kbb);
    }
    run_gemm(stream, true, false, false, qb, CC, TC, CD, kbb, CC, TC, CD,
             scores, CT, (long)CH * TT, TT, CT, CT, CD, CN * CH, CH);
    softmax_kernel<true><<<dim3(CT, CH, CN), 256, 0, stream>>>(
        scores, sb, nullptr, CT, CT);
    run_gemm(stream, false, false, true, sb, CT, (long)CH * TT, TT, vbb, CC, TC, CD,
             ohb, CC, TC, CD, CT, CD, CT, CN * CH, CH);
    run_gemm(stream, false, true, false, ohb, CC, 0, 0, bWo_t, CC, 0, 0,
             y, CC, 0, 0, CN * CT, CC, CC, 1, 1);

    // ---- MLP ----
    ln_kernel<<<CN * CT, 256, 0, stream>>>(y, ln2_w + (size_t)l * CC, xnb, CC);
    run_gemm(stream, false, false, false, xnb, CC, 0, 0, bW1, CFF, 0, 0,
             ffh, CFF, 0, 0, CN * CT, CFF, CC, 1, 1);
    gelu_kernel<<<(unsigned)((NTF + 255) / 256), 256, 0, stream>>>(ffh, ffhb, NTF);
    run_gemm(stream, false, true, false, ffhb, CFF, 0, 0, bW2, CC, 0, 0,
             y, CC, 0, 0, CN * CT, CC, CFF, 1, 1);
  }

  // final LN + logits + loss
  ln_kernel<<<CN * CT, 256, 0, stream>>>(y, lnf_w, xnb, CC);
  run_gemm(stream, false, false, false, xnb, CC, 0, 0, wob, CV, 0, 0,
           logits, CV, 0, 0, CN * CT, CV, CC, 1, 1);
  init_accum<<<1, 1, 0, stream>>>(accum);
  loss_kernel<<<dim3(CT, CN), 256, 0, stream>>>(logits, targets, target_lengths, accum);
  finalize_kernel<<<1, 1, 0, stream>>>(accum, (float*)d_out);
}
